// PieceWiseCNN_53188874993739
// MI455X (gfx1250) — matrix operand order repaired and checked by reading, not compiled, not run on hardware
//
#include <hip/hip_runtime.h>
#include <hip/hip_bf16.h>
#include <math.h>

// Problem constants (from reference)
#define BB   128
#define LL   512
#define DE   384      // fc1 in/out, conv in-channels
#define DP   192      // fc2 in
#define NF   256      // conv filters
#define KC   1152     // conv GEMM K = 3*384

// LDS padded strides (halfwords). 392*2B=784B -> 196 dwords, 196%64=4:
// the 16 A-fragment rows land on distinct bank groups for ds_read_b128.
#define SXE  392
#define SXP  200

typedef __attribute__((ext_vector_type(16))) __bf16 v16bf;
typedef __attribute__((ext_vector_type(8)))  __bf16 v8bf;
typedef __attribute__((ext_vector_type(8)))  float  v8f;

__device__ __forceinline__ unsigned short f32_to_bf16(float f) {
  union { float f; unsigned u; } v; v.f = f;
  unsigned r = v.u + 0x7FFFu + ((v.u >> 16) & 1u);   // round-to-nearest-even
  return (unsigned short)(r >> 16);
}
// Monotonic float -> uint key so unsigned atomicMax == float max (handles negatives)
__device__ __forceinline__ unsigned f32_to_key(float f) {
  unsigned u = __float_as_uint(f);
  return (u & 0x80000000u) ? ~u : (u | 0x80000000u);
}
__device__ __forceinline__ float key_to_f32(unsigned k) {
  unsigned u = (k & 0x80000000u) ? (k & 0x7FFFFFFFu) : ~k;
  return __uint_as_float(u);
}

// A-fragment: 16x32 bf16 rows from LDS (row = lane&15), two 16B loads.
__device__ __forceinline__ v16bf load_a_frag(const unsigned short* row, int k0, int laneH) {
  v16bf a;
  ((v8bf*)&a)[0] = *(const v8bf*)(row + k0 +     laneH * 16);
  ((v8bf*)&a)[1] = *(const v8bf*)(row + k0 + 8 + laneH * 16);
  return a;
}

// ---------------------------------------------------------------------------
// Kernel 0: weight conversion/repack + pooling-key init
// ---------------------------------------------------------------------------
__global__ void pw_prep(const float* __restrict__ W1, const float* __restrict__ W2,
                        const float* __restrict__ Wc,
                        unsigned short* __restrict__ w1b, unsigned short* __restrict__ w2b,
                        unsigned short* __restrict__ wcb, unsigned* __restrict__ keys) {
  int i = blockIdx.x * 256 + threadIdx.x;
  if (i < DE * DE) w1b[i] = f32_to_bf16(W1[i]);                 // [e][d], K-contig
  if (i < DE * DP) w2b[i] = f32_to_bf16(W2[i]);                 // [e][d], K-contig
  if (i < NF * KC) {                                            // [o][t][c] <- Wc[o][c][t]
    int o = i / KC, rem = i % KC, t = rem / DE, c = rem % DE;
    wcb[i] = f32_to_bf16(Wc[(size_t)o * KC + c * 3 + t]);
  }
  if (i < BB * NF * 3) keys[i] = 0u;                            // below key(-inf)
}

// ---------------------------------------------------------------------------
// Kernel 1: gated fusion.  One block = 32 rows of (B*L) x full N=384.
// 8 waves; each wave: 2 M-tiles x 3 N-tiles, B-fragment shared across M-tiles.
// GEMM1 (K=384) then GEMM2 (K=192), both bf16 WMMA, f32 accumulate.
// ---------------------------------------------------------------------------
__global__ void __launch_bounds__(256)
pw_gate(const float* __restrict__ Xp, const float* __restrict__ Xe,
        const unsigned short* __restrict__ w1b, const float* __restrict__ b1,
        const unsigned short* __restrict__ w2b, const float* __restrict__ b2,
        unsigned short* __restrict__ Xw) {
  __shared__ unsigned short XeT[32 * SXE];    // 25088 B
  __shared__ unsigned short XpT[32 * SXP];    // 12800 B

  const int tid  = threadIdx.x;
  const int row0 = blockIdx.x * 32;           // row = b*512 + l

  for (int i = tid; i < 32 * DE; i += 256) {
    int r = i / DE, c = i % DE;
    XeT[r * SXE + c] = f32_to_bf16(Xe[(size_t)(row0 + r) * DE + c]);
  }
  for (int i = tid; i < 32 * DP; i += 256) {
    int r = i / DP, c = i % DP;
    XpT[r * SXP + c] = f32_to_bf16(Xp[(size_t)(row0 + r) * DP + c]);
  }
  __syncthreads();

  const int wave = tid >> 5, lane = tid & 31;
  const int laneN = lane & 15, laneH = lane >> 4;
  const int kb16 = laneH * 16;
  const int n0   = wave * 48;                 // this wave's 3 N-tiles
  const unsigned short* arow0E = XeT + (size_t)laneN * SXE;
  const unsigned short* arow1E = XeT + (size_t)(16 + laneN) * SXE;
  const unsigned short* arow0P = XpT + (size_t)laneN * SXP;
  const unsigned short* arow1P = XpT + (size_t)(16 + laneN) * SXP;

  // ---- GEMM1: A = sigmoid(Xe @ W1^T + b1) ----
  v8f accA[2][3];
#pragma unroll
  for (int j = 0; j < 3; ++j) {
    const float bias = b1[n0 + j * 16 + laneN];
#pragma unroll
    for (int r = 0; r < 8; ++r) { accA[0][j][r] = bias; accA[1][j][r] = bias; }
  }
  for (int k0 = 0; k0 < DE; k0 += 32) {
    v16bf a0 = load_a_frag(arow0E, k0, laneH);
    v16bf a1 = load_a_frag(arow1E, k0, laneH);
#pragma unroll
    for (int j = 0; j < 3; ++j) {
      const int n = n0 + j * 16 + laneN;
      const unsigned short* bp = w1b + (size_t)n * DE + k0 + kb16;
      if (k0 + 32 < DE) __builtin_prefetch(bp + 32, 0, 1);  // global_prefetch_b8
      v16bf b = *(const v16bf*)bp;
      accA[0][j] = __builtin_amdgcn_wmma_f32_16x16x32_bf16(false, a0, false, b,
                                                           (short)0, accA[0][j], false, false);
      accA[1][j] = __builtin_amdgcn_wmma_f32_16x16x32_bf16(false, a1, false, b,
                                                           (short)0, accA[1][j], false, false);
    }
  }
#pragma unroll
  for (int mt = 0; mt < 2; ++mt)
#pragma unroll
    for (int j = 0; j < 3; ++j)
#pragma unroll
      for (int r = 0; r < 8; ++r)
        accA[mt][j][r] = 1.0f / (1.0f + __expf(-accA[mt][j][r]));

  // ---- GEMM2: T = tanh(Xp @ W2^T + b2) ----
  v8f accT[2][3];
#pragma unroll
  for (int j = 0; j < 3; ++j) {
    const float bias = b2[n0 + j * 16 + laneN];
#pragma unroll
    for (int r = 0; r < 8; ++r) { accT[0][j][r] = bias; accT[1][j][r] = bias; }
  }
  for (int k0 = 0; k0 < DP; k0 += 32) {
    v16bf a0 = load_a_frag(arow0P, k0, laneH);
    v16bf a1 = load_a_frag(arow1P, k0, laneH);
#pragma unroll
    for (int j = 0; j < 3; ++j) {
      const int n = n0 + j * 16 + laneN;
      v16bf b = *(const v16bf*)(w2b + (size_t)n * DP + k0 + kb16);
      accT[0][j] = __builtin_amdgcn_wmma_f32_16x16x32_bf16(false, a0, false, b,
                                                           (short)0, accT[0][j], false, false);
      accT[1][j] = __builtin_amdgcn_wmma_f32_16x16x32_bf16(false, a1, false, b,
                                                           (short)0, accT[1][j], false, false);
    }
  }

  // ---- X = A*Xe + (1-A)*tanh(...); keep in regs (overwrite accA) ----
#pragma unroll
  for (int mt = 0; mt < 2; ++mt)
#pragma unroll
    for (int j = 0; j < 3; ++j) {
      const int n = n0 + j * 16 + laneN;
#pragma unroll
      for (int r = 0; r < 8; ++r) {
        const int m = mt * 16 + r + laneH * 8;
        const float xe = Xe[(size_t)(row0 + m) * DE + n];  // exact f32 gate term
        const float a  = accA[mt][j][r];
        accA[mt][j][r] = a * xe + (1.0f - a) * tanhf(accT[mt][j][r]);
      }
    }

  // ---- stage result tile back into XeT, then coalesced b128 stores ----
  __syncthreads();                            // all waves done reading XeT
#pragma unroll
  for (int mt = 0; mt < 2; ++mt)
#pragma unroll
    for (int j = 0; j < 3; ++j) {
      const int n = n0 + j * 16 + laneN;
#pragma unroll
      for (int r = 0; r < 8; ++r) {
        const int m = mt * 16 + r + laneH * 8;
        XeT[m * SXE + n] = f32_to_bf16(accA[mt][j][r]);
      }
    }
  __syncthreads();
  for (int i = tid * 8; i < 32 * DE; i += 256 * 8) {
    const int r = i / DE, c = i % DE;         // c is a multiple of 8 (384 % 8 == 0)
    *(uint4*)(Xw + (size_t)(row0 + r) * DE + c) = *(const uint4*)(XeT + r * SXE + c);
  }
}

// ---------------------------------------------------------------------------
// Kernel 2: conv1d (pad=1, K=3) as WMMA GEMM (K=1152) + piecewise max-pool.
// One block = one batch's 32-row L-strip x all 256 filters.
// 8 waves; each wave: 2 M-tiles x 2 N-tiles, B shared across M-tiles.
// ---------------------------------------------------------------------------
__global__ void __launch_bounds__(256)
pw_conv_pool(const unsigned short* __restrict__ Xw, const unsigned short* __restrict__ wcb,
             const float* __restrict__ bc, const int* __restrict__ Xm,
             const float* __restrict__ mask_emb, unsigned* __restrict__ keys) {
  __shared__ unsigned short Xs[34 * SXE];     // rows l0-1 .. l0+32 (halo), zero padded
  __shared__ float mrow[32][3];

  const int tid = threadIdx.x;
  const int b   = blockIdx.x >> 4;            // 16 strips of 32 per batch
  const int l0  = (blockIdx.x & 15) * 32;
  const size_t base = (size_t)b * LL * DE;

  for (int i = tid; i < 34 * DE; i += 256) {
    int r = i / DE, c = i % DE;
    int l = l0 - 1 + r;
    unsigned short v = 0;                     // zero padding outside [0,512)
    if (l >= 0 && l < LL) v = Xw[base + (size_t)l * DE + c];
    Xs[r * SXE + c] = v;
  }
  if (tid < 32) {
    int mk = Xm[b * LL + l0 + tid];
    mrow[tid][0] = mask_emb[mk * 3 + 0];
    mrow[tid][1] = mask_emb[mk * 3 + 1];
    mrow[tid][2] = mask_emb[mk * 3 + 2];
  }
  __syncthreads();

  const int wave = tid >> 5, lane = tid & 31;
  const int laneN = lane & 15, laneH = lane >> 4;
  const int kb16 = laneH * 16;
  const int n0   = wave * 32;                 // this wave's 2 N-tiles

  v8f acc[2][2];
#pragma unroll
  for (int j = 0; j < 2; ++j) {
    const float bias = bc[n0 + j * 16 + laneN];
#pragma unroll
    for (int r = 0; r < 8; ++r) { acc[0][j][r] = bias; acc[1][j][r] = bias; }
  }

#pragma unroll
  for (int t = 0; t < 3; ++t) {
    // output row m uses X row (l0+m+t-1) -> LDS row (m+t); A-matrix row = laneN
    const unsigned short* arow0 = Xs + (size_t)(laneN + t) * SXE;
    const unsigned short* arow1 = Xs + (size_t)(laneN + 16 + t) * SXE;
    for (int kc = 0; kc < DE; kc += 32) {
      v16bf a0 = load_a_frag(arow0, kc, laneH);
      v16bf a1 = load_a_frag(arow1, kc, laneH);
#pragma unroll
      for (int j = 0; j < 2; ++j) {
        const int n = n0 + j * 16 + laneN;
        const unsigned short* bp = wcb + (size_t)n * KC + t * DE + kc + kb16;
        if (kc + 32 < DE) __builtin_prefetch(bp + 32, 0, 1);
        v16bf bf = *(const v16bf*)bp;
        acc[0][j] = __builtin_amdgcn_wmma_f32_16x16x32_bf16(false, a0, false, bf,
                                                            (short)0, acc[0][j], false, false);
        acc[1][j] = __builtin_amdgcn_wmma_f32_16x16x32_bf16(false, a1, false, bf,
                                                            (short)0, acc[1][j], false, false);
      }
    }
  }

  // ---- piecewise max over this strip's 32 L-rows, fold into global keys ----
#pragma unroll
  for (int j = 0; j < 2; ++j) {
    const int n = n0 + j * 16 + laneN;
#pragma unroll
    for (int jj = 0; jj < 3; ++jj) {
      float mx = -3.402823466e38f;
#pragma unroll
      for (int mt = 0; mt < 2; ++mt)
#pragma unroll
        for (int r = 0; r < 8; ++r) {
          const int m = mt * 16 + r + laneH * 8;
          mx = fmaxf(mx, mrow[m][jj] * acc[mt][j][r]);
        }
      mx = fmaxf(mx, __shfl_xor(mx, 16, 32));   // combine the two 8-row halves
      if (laneH == 0)
        atomicMax(&keys[((size_t)b * NF + n) * 3 + jj], f32_to_key(mx));
    }
  }
}

// ---------------------------------------------------------------------------
// Kernel 3: decode keys, tanh, write output (layout b*768 + h*3 + j)
// ---------------------------------------------------------------------------
__global__ void pw_finalize(const unsigned* __restrict__ keys, float* __restrict__ out) {
  int i = blockIdx.x * 256 + threadIdx.x;
  if (i < BB * NF * 3) out[i] = tanhf(key_to_f32(keys[i]));
}

// ---------------------------------------------------------------------------
extern "C" void kernel_launch(void* const* d_in, const int* in_sizes, int n_in,
                              void* d_out, int out_size, void* d_ws, size_t ws_size,
                              hipStream_t stream) {
  (void)in_sizes; (void)n_in; (void)out_size; (void)ws_size;
  const float* Xp = (const float*)d_in[0];
  const float* Xe = (const float*)d_in[1];
  const int*   Xm = (const int*)d_in[2];
  const float* W1 = (const float*)d_in[3];
  const float* b1 = (const float*)d_in[4];
  const float* W2 = (const float*)d_in[5];
  const float* b2 = (const float*)d_in[6];
  const float* Wc = (const float*)d_in[7];
  const float* bc = (const float*)d_in[8];
  const float* me = (const float*)d_in[9];

  // Workspace layout (all offsets 32B-aligned):
  //   Xw  : 65536*384 bf16  = 50331648 B   (gated activations)
  //   w1b : 384*384   bf16  =   294912 B
  //   w2b : 384*192   bf16  =   147456 B
  //   wcb : 256*1152  bf16  =   589824 B   (conv weights repacked [o][t][c])
  //   keys: 128*256*3 u32   =   393216 B   (~49.4 MB total)
  char* ws = (char*)d_ws;
  unsigned short* Xw   = (unsigned short*)(ws);
  unsigned short* w1b  = (unsigned short*)(ws + 50331648);
  unsigned short* w2b  = (unsigned short*)(ws + 50331648 + 294912);
  unsigned short* wcb  = (unsigned short*)(ws + 50331648 + 294912 + 147456);
  unsigned*       keys = (unsigned*)(ws + 50331648 + 294912 + 147456 + 589824);

  pw_prep<<<(NF * KC + 255) / 256, 256, 0, stream>>>(W1, W2, Wc, w1b, w2b, wcb, keys);
  pw_gate<<<(BB * LL) / 32, 256, 0, stream>>>(Xp, Xe, w1b, b1, w2b, b2, Xw);
  pw_conv_pool<<<(BB * LL) / 32, 256, 0, stream>>>(Xw, wcb, bc, Xm, me, keys);
  pw_finalize<<<(BB * NF * 3 + 255) / 256, 256, 0, stream>>>(keys, (float*)d_out);
}
